// NanoGPT_46308337386425
// MI455X (gfx1250) — matrix operand order repaired and checked by reading, not compiled, not run on hardware
//
#include <hip/hip_runtime.h>

// ---------------- NanoGPT forward on gfx1250 (CDNA5, wave32, WMMA bf16) ----------
#define NG_B   4
#define NG_T   1024
#define NG_D   768
#define NG_H   12
#define NG_HD  64
#define NG_L   8
#define NG_V   32000
#define NG_M   (NG_B * NG_T)   // 4096 rows

#define USE_ASYNC_LDS 1

typedef __attribute__((ext_vector_type(16))) __bf16 v16bf;
typedef __attribute__((ext_vector_type(8)))  __bf16 v8bf;
typedef __attribute__((ext_vector_type(8)))  float  v8f;
typedef unsigned short u16;

union U16frag { v16bf v; v8bf h2[2]; };

#if __has_builtin(__builtin_amdgcn_cvt_pk_bf16_f32)
typedef __attribute__((ext_vector_type(2))) __bf16 v2bf;
__device__ __forceinline__ unsigned pack2bf(float lo, float hi) {
    union { v2bf v; unsigned u; } c;
    c.v = __builtin_amdgcn_cvt_pk_bf16_f32(lo, hi);
    return c.u;
}
#else
__device__ __forceinline__ unsigned pack2bf(float lo, float hi) {
    union { float f; unsigned u; } a, b; a.f = lo; b.f = hi;
    unsigned ua = a.u + (0x7fffu + ((a.u >> 16) & 1u));   // RNE
    unsigned ub = b.u + (0x7fffu + ((b.u >> 16) & 1u));
    return (ua >> 16) | (ub & 0xffff0000u);
}
#endif

__device__ __forceinline__ uint2 pack4bf(float4 v) {
    uint2 o; o.x = pack2bf(v.x, v.y); o.y = pack2bf(v.z, v.w); return o;
}

__device__ __forceinline__ u16 f2bf(float f) {
    return (u16)(pack2bf(f, 0.f) & 0xffffu);
}

__device__ __forceinline__ float gelu_tanh(float x) {
    const float c = 0.7978845608028654f; // sqrt(2/pi)
    float t = tanhf(c * (x + 0.044715f * x * x * x));
    return 0.5f * x * (1.0f + t);
}

__device__ __forceinline__ v8f wmma_bf16(v16bf a, v16bf b, v8f c) {
    return __builtin_amdgcn_wmma_f32_16x16x32_bf16(false, a, false, b, (short)0, c,
                                                   false, false);
}

// 16-byte global -> LDS copy. Async path uses the CDNA5 async-to-LDS DMA
// (tracked by ASYNCcnt); fallback is a plain VGPR round-trip.
__device__ __forceinline__ void copy16_g2l(u16* lds_ptr, const u16* gptr) {
#if USE_ASYNC_LDS
    asm volatile("global_load_async_to_lds_b128 %0, %1, off"
                 :: "v"((unsigned)(size_t)lds_ptr), "v"(gptr)
                 : "memory");
#else
    *(uint4*)lds_ptr = *(const uint4*)gptr;
#endif
}
__device__ __forceinline__ void async_fence_all() {
#if USE_ASYNC_LDS
    asm volatile("s_wait_asynccnt 0" ::: "memory");
#endif
}

// ---------------------------------------------------------------------------
// Embedding (fp32 residual stream)
// ---------------------------------------------------------------------------
__global__ __launch_bounds__(256) void ng_embed(const int* __restrict__ idx,
        const float* __restrict__ tok, const float* __restrict__ pos,
        float* __restrict__ x) {
    int row = blockIdx.x;
    int t   = row & (NG_T - 1);
    int token = idx[row];
    const float* tp = tok + (size_t)token * NG_D;
    const float* pp = pos + (size_t)t * NG_D;
    float* xp = x + (size_t)row * NG_D;
    for (int j = threadIdx.x; j < NG_D; j += 256) xp[j] = tp[j] + pp[j];
}

// ---------------------------------------------------------------------------
// LayerNorm (fp32 in, bf16 out -> feeds WMMA GEMMs directly)
// ---------------------------------------------------------------------------
__global__ __launch_bounds__(256) void ng_ln(const float* __restrict__ x,
        const float* __restrict__ w, const float* __restrict__ b,
        u16* __restrict__ out) {
    __shared__ float red[256];
    int row = blockIdx.x, tid = threadIdx.x;
    const float* xr = x + (size_t)row * NG_D;
    float s = 0.f;
    for (int j = tid; j < NG_D; j += 256) s += xr[j];
    red[tid] = s; __syncthreads();
    for (int off = 128; off > 0; off >>= 1) {
        if (tid < off) red[tid] += red[tid + off];
        __syncthreads();
    }
    float mu = red[0] * (1.0f / NG_D);
    __syncthreads();
    float v = 0.f;
    for (int j = tid; j < NG_D; j += 256) { float d = xr[j] - mu; v += d * d; }
    red[tid] = v; __syncthreads();
    for (int off = 128; off > 0; off >>= 1) {
        if (tid < off) red[tid] += red[tid + off];
        __syncthreads();
    }
    float inv = rsqrtf(red[0] * (1.0f / NG_D) + 1e-5f);
    u16* op = out + (size_t)row * NG_D;
    for (int j = tid; j < NG_D; j += 256)
        op[j] = f2bf((xr[j] - mu) * inv * w[j] + b[j]);
}

// ---------------------------------------------------------------------------
// Elementwise fp32 -> bf16 convert (tok_emb for tied lm_head)
// ---------------------------------------------------------------------------
__global__ __launch_bounds__(256) void ng_conv(const float* __restrict__ src,
        u16* __restrict__ dst, int n4) {
    int i = blockIdx.x * 256 + threadIdx.x;
    if (i < n4) {
        float4 v = *(const float4*)(src + (size_t)i * 4);
        *(uint2*)(dst + (size_t)i * 4) = pack4bf(v);
    }
}

// ---------------------------------------------------------------------------
// Weight transpose-convert: W[K,N] fp32 -> Wt[N,K] bf16 (64x64 LDS tiles)
// ---------------------------------------------------------------------------
__global__ __launch_bounds__(256) void ng_wconv_t(const float* __restrict__ W,
        u16* __restrict__ Wt, int K, int N) {
    __shared__ u16 tile[64][68];
    int k0 = blockIdx.x * 64, n0 = blockIdx.y * 64;
    int tid = threadIdx.x;
    int tr  = tid >> 4;               // + i*16
    int tc4 = (tid & 15) * 4;
    #pragma unroll
    for (int i = 0; i < 4; ++i) {
        int kk = tr + i * 16;
        float4 v = *(const float4*)(W + (size_t)(k0 + kk) * N + n0 + tc4);
        *(uint2*)&tile[kk][tc4] = pack4bf(v);
    }
    __syncthreads();
    #pragma unroll
    for (int i = 0; i < 4; ++i) {
        int nn = tr + i * 16;
        uint2 o;
        o.x = (unsigned)tile[tc4 + 0][nn] | ((unsigned)tile[tc4 + 1][nn] << 16);
        o.y = (unsigned)tile[tc4 + 2][nn] | ((unsigned)tile[tc4 + 3][nn] << 16);
        *(uint2*)(Wt + (size_t)(n0 + nn) * K + k0 + tc4) = o;
    }
}

// ---------------------------------------------------------------------------
// GEMM: C[M,N] = act(A @ Bt^T + bias) (+ resid)
//   A : bf16 [M,K], Bt : bf16 [N,K], C : fp32 or bf16 (out_bf16)
// Tile 128x128x64, double-buffered LDS (64KB), async-to-LDS staging,
// 16 WMMAs per k-block per wave-group.
// ---------------------------------------------------------------------------
__global__ __launch_bounds__(256) void ng_gemm(const u16* __restrict__ A,
        const u16* __restrict__ Bt, const float* __restrict__ bias,
        const float* __restrict__ resid, void* __restrict__ Cout,
        int M, int N, int K, int act, int out_bf16) {
    __shared__ alignas(32) u16 sA[2][128 * 64];
    __shared__ alignas(32) u16 sB[2][128 * 64];

    const int tid  = threadIdx.x;
    const int lane = tid & 31;
    const int wm = (tid >> 5) & 3, wn = tid >> 7;
    const int r = lane & 15, hv = lane >> 4;
    const int m0 = blockIdx.x * 128, n0 = blockIdx.y * 128;

    const int srow0 = tid >> 3;          // + i*32
    const int scol8 = (tid & 7) * 8;
    const u16* pa = A  + (size_t)(m0 + srow0) * K + scol8;
    const u16* pb = Bt + (size_t)(n0 + srow0) * K + scol8;

    auto stage = [&](int buf) {
        #pragma unroll
        for (int i = 0; i < 4; ++i)
            copy16_g2l(&sA[buf][(srow0 + i * 32) * 64 + scol8], pa + (size_t)i * 32 * K);
        #pragma unroll
        for (int i = 0; i < 4; ++i)
            copy16_g2l(&sB[buf][(srow0 + i * 32) * 64 + scol8], pb + (size_t)i * 32 * K);
        pa += 64; pb += 64;
    };

    v8f acc[8] = {{}, {}, {}, {}, {}, {}, {}, {}};

    stage(0);
    const int nkb = K / 64;
    for (int kb = 0; kb < nkb; ++kb) {
        async_fence_all();
        __syncthreads();
        const int buf = kb & 1;
        if (kb + 1 < nkb) stage(buf ^ 1);      // DMA next tile during WMMAs
        #pragma unroll
        for (int ks = 0; ks < 2; ++ks) {
            U16frag af[2];
            #pragma unroll
            for (int am = 0; am < 2; ++am) {
                const u16* par = &sA[buf][(wm * 32 + am * 16 + r) * 64 + ks * 32];
                af[am].h2[0] = *(const v8bf*)(par + hv * 16);
                af[am].h2[1] = *(const v8bf*)(par + hv * 16 + 8);
            }
            #pragma unroll
            for (int nt = 0; nt < 4; ++nt) {
                v16bf bf = *(const v16bf*)(&sB[buf][(wn * 64 + nt * 16 + r) * 64
                                                    + ks * 32 + hv * 16]);
                acc[0 * 4 + nt] = wmma_bf16(af[0].v, bf, acc[0 * 4 + nt]);
                acc[1 * 4 + nt] = wmma_bf16(af[1].v, bf, acc[1 * 4 + nt]);
            }
        }
    }

    #pragma unroll
    for (int am = 0; am < 2; ++am) {
        #pragma unroll
        for (int nt = 0; nt < 4; ++nt) {
            #pragma unroll
            for (int i = 0; i < 8; ++i) {
                int row = m0 + wm * 32 + am * 16 + hv * 8 + i;
                int col = n0 + wn * 64 + nt * 16 + r;
                float v = acc[am * 4 + nt][i];
                if (bias) v += bias[col];
                if (act == 1) v = gelu_tanh(v);
                size_t o = (size_t)row * N + col;
                if (out_bf16) {
                    ((u16*)Cout)[o] = f2bf(v);
                } else {
                    if (resid) v += resid[o];
                    ((float*)Cout)[o] = v;
                }
            }
        }
    }
}

// ---------------------------------------------------------------------------
// Flash attention over bf16 qkv, one (b,h,64-query) tile per block.
// ---------------------------------------------------------------------------
__global__ __launch_bounds__(256) void ng_attn(const u16* __restrict__ qkv,
                                               u16* __restrict__ y) {
    __shared__ alignas(32) u16 sQ[64 * 64];
    __shared__ alignas(32) u16 sK[64 * 64];
    __shared__ alignas(32) u16 sVt[64 * 64];
    __shared__ alignas(32) u16 sP[64 * 64];
    __shared__ float redmax[2][64];
    __shared__ float redsum[2][64];
    __shared__ float m_run[64];
    __shared__ float l_run[64];

    const int S3D = 3 * NG_D;
    int qt = blockIdx.x;
    int bh = blockIdx.y;
    int b  = bh / NG_H, h = bh % NG_H;
    int q0 = qt * 64;

    int tid = threadIdx.x;
    int lane = tid & 31;
    int wm = (tid >> 5) & 3, wn = tid >> 7;
    int r = lane & 15, hv = lane >> 4;

    const int trow0 = tid >> 3;          // + i*32
    const int tcol8 = (tid & 7) * 8;

    #pragma unroll
    for (int i = 0; i < 2; ++i) {
        int rw = trow0 + i * 32;
        copy16_g2l(&sQ[rw * 64 + tcol8],
                   qkv + (size_t)(b * NG_T + q0 + rw) * S3D + h * NG_HD + tcol8);
    }
    if (tid < 64) { m_run[tid] = -3.0e38f; l_run[tid] = 0.f; }
    v8f o0 = {}, o1 = {};

    const int vkey = (tid >> 4) * 4;
    const int vd   = (tid & 15) * 4;

    for (int st = 0; st <= qt; ++st) {
        int s0 = st * 64;
        #pragma unroll
        for (int i = 0; i < 2; ++i) {
            int rw = trow0 + i * 32;
            copy16_g2l(&sK[rw * 64 + tcol8],
                       qkv + (size_t)(b * NG_T + s0 + rw) * S3D + NG_D
                           + h * NG_HD + tcol8);
        }
        {
            uint2 w0 = *(const uint2*)(qkv + (size_t)(b * NG_T + s0 + vkey + 0) * S3D
                                       + 2 * NG_D + h * NG_HD + vd);
            uint2 w1 = *(const uint2*)(qkv + (size_t)(b * NG_T + s0 + vkey + 1) * S3D
                                       + 2 * NG_D + h * NG_HD + vd);
            uint2 w2 = *(const uint2*)(qkv + (size_t)(b * NG_T + s0 + vkey + 2) * S3D
                                       + 2 * NG_D + h * NG_HD + vd);
            uint2 w3 = *(const uint2*)(qkv + (size_t)(b * NG_T + s0 + vkey + 3) * S3D
                                       + 2 * NG_D + h * NG_HD + vd);
            #pragma unroll
            for (int j = 0; j < 4; ++j) {
                unsigned e0 = (((j < 2) ? w0.x : w0.y) >> ((j & 1) * 16)) & 0xffffu;
                unsigned e1 = (((j < 2) ? w1.x : w1.y) >> ((j & 1) * 16)) & 0xffffu;
                unsigned e2 = (((j < 2) ? w2.x : w2.y) >> ((j & 1) * 16)) & 0xffffu;
                unsigned e3 = (((j < 2) ? w3.x : w3.y) >> ((j & 1) * 16)) & 0xffffu;
                uint2 o; o.x = e0 | (e1 << 16); o.y = e2 | (e3 << 16);
                *(uint2*)&sVt[(vd + j) * 64 + vkey] = o;
            }
        }
        async_fence_all();
        __syncthreads();

        v8f sa0 = {}, sa1 = {};
        #pragma unroll
        for (int ks = 0; ks < 2; ++ks) {
            const u16* pa = &sQ[(wm * 16 + r) * 64 + ks * 32];
            U16frag af;
            af.h2[0] = *(const v8bf*)(pa + hv * 16);
            af.h2[1] = *(const v8bf*)(pa + hv * 16 + 8);
            v16bf b0 = *(const v16bf*)(&sK[(wn * 32 + r) * 64 + ks * 32 + hv * 16]);
            v16bf b1 = *(const v16bf*)(&sK[(wn * 32 + 16 + r) * 64 + ks * 32 + hv * 16]);
            sa0 = wmma_bf16(af.v, b0, sa0);
            sa1 = wmma_bf16(af.v, b1, sa1);
        }

        const float NEG = -3.0e38f;
        int cg0 = s0 + wn * 32 + r;
        int cg1 = cg0 + 16;
        #pragma unroll
        for (int i = 0; i < 8; ++i) {
            int rg = q0 + wm * 16 + hv * 8 + i;
            float v0 = (cg0 <= rg) ? sa0[i] * 0.125f : NEG;
            float v1 = (cg1 <= rg) ? sa1[i] * 0.125f : NEG;
            sa0[i] = v0; sa1[i] = v1;
            float mx = fmaxf(v0, v1);
            mx = fmaxf(mx, __shfl_xor(mx, 1, 32));
            mx = fmaxf(mx, __shfl_xor(mx, 2, 32));
            mx = fmaxf(mx, __shfl_xor(mx, 4, 32));
            mx = fmaxf(mx, __shfl_xor(mx, 8, 32));
            if (r == 0) redmax[wn][wm * 16 + hv * 8 + i] = mx;
        }
        __syncthreads();

        #pragma unroll
        for (int i = 0; i < 8; ++i) {
            int rl = wm * 16 + hv * 8 + i;
            float m_old = m_run[rl];
            float m_new = fmaxf(m_old, fmaxf(redmax[0][rl], redmax[1][rl]));
            float sc = __expf(m_old - m_new);
            float p0 = __expf(sa0[i] - m_new);
            float p1 = __expf(sa1[i] - m_new);
            sP[rl * 64 + wn * 32 + r]      = f2bf(p0);
            sP[rl * 64 + wn * 32 + 16 + r] = f2bf(p1);
            o0[i] *= sc; o1[i] *= sc;
            float ps = p0 + p1;
            ps += __shfl_xor(ps, 1, 32);
            ps += __shfl_xor(ps, 2, 32);
            ps += __shfl_xor(ps, 4, 32);
            ps += __shfl_xor(ps, 8, 32);
            if (r == 0) redsum[wn][rl] = ps;
        }
        __syncthreads();

        if (tid < 64) {
            float m_old = m_run[tid];
            float m_new = fmaxf(m_old, fmaxf(redmax[0][tid], redmax[1][tid]));
            l_run[tid] = l_run[tid] * __expf(m_old - m_new)
                       + redsum[0][tid] + redsum[1][tid];
            m_run[tid] = m_new;
        }

        #pragma unroll
        for (int ks = 0; ks < 2; ++ks) {
            const u16* pa = &sP[(wm * 16 + r) * 64 + ks * 32];
            U16frag af;
            af.h2[0] = *(const v8bf*)(pa + hv * 16);
            af.h2[1] = *(const v8bf*)(pa + hv * 16 + 8);
            v16bf b0 = *(const v16bf*)(&sVt[(wn * 32 + r) * 64 + ks * 32 + hv * 16]);
            v16bf b1 = *(const v16bf*)(&sVt[(wn * 32 + 16 + r) * 64 + ks * 32 + hv * 16]);
            o0 = wmma_bf16(af.v, b0, o0);
            o1 = wmma_bf16(af.v, b1, o1);
        }
        __syncthreads();
    }

    #pragma unroll
    for (int i = 0; i < 8; ++i) {
        int rl = wm * 16 + hv * 8 + i;
        float inv = 1.0f / l_run[rl];
        size_t base = (size_t)(b * NG_T + q0 + rl) * NG_D + h * NG_HD + wn * 32 + r;
        y[base]      = f2bf(o0[i] * inv);
        y[base + 16] = f2bf(o1[i] * inv);
    }
}

// ---------------------------------------------------------------------------
// Launch. d_in order = insertion order of setup_inputs().
// ---------------------------------------------------------------------------
extern "C" void kernel_launch(void* const* d_in, const int* in_sizes, int n_in,
                              void* d_out, int out_size, void* d_ws, size_t ws_size,
                              hipStream_t stream) {
    (void)in_sizes; (void)n_in; (void)out_size; (void)ws_size;
    const int M = NG_M;
    const int D = NG_D;

    const int*   idx = (const int*)d_in[0];
    const float* tok = (const float*)d_in[1];
    const float* pos = (const float*)d_in[2];
    const float* lnf_w = (const float*)d_in[3 + 12 * NG_L];
    const float* lnf_b = (const float*)d_in[3 + 12 * NG_L + 1];

    float* x     = (float*)d_ws;                       // [M,D]  fp32 residual
    u16*   h16   = (u16*)(x + (size_t)M * D);          // [M,D]  bf16
    u16*   qkv16 = h16 + (size_t)M * D;                // [M,3D] bf16
    u16*   yat16 = qkv16 + (size_t)M * 3 * D;          // [M,D]  bf16
    u16*   fcb16 = yat16 + (size_t)M * D;              // [M,4D] bf16
    u16*   tok16 = fcb16 + (size_t)M * 4 * D;          // [V,D]  bf16
    u16*   wq    = tok16 + (size_t)NG_V * D;           // per-layer weight scratch
    u16*   wp    = wq + (size_t)3 * D * D;
    u16*   wf    = wp + (size_t)D * D;
    u16*   w2    = wf + (size_t)4 * D * D;

    ng_embed<<<M, 256, 0, stream>>>(idx, tok, pos, x);
    ng_conv<<<(NG_V * D / 4 + 255) / 256, 256, 0, stream>>>(tok, tok16, NG_V * D / 4);

    for (int l = 0; l < NG_L; ++l) {
        const int base = 3 + l * 12;
        const float* ln1_w = (const float*)d_in[base + 0];
        const float* ln1_b = (const float*)d_in[base + 1];
        const float* Wqkv  = (const float*)d_in[base + 2];
        const float* bqkv  = (const float*)d_in[base + 3];
        const float* Wproj = (const float*)d_in[base + 4];
        const float* bproj = (const float*)d_in[base + 5];
        const float* ln2_w = (const float*)d_in[base + 6];
        const float* ln2_b = (const float*)d_in[base + 7];
        const float* Wfc   = (const float*)d_in[base + 8];
        const float* bfc   = (const float*)d_in[base + 9];
        const float* Wfc2  = (const float*)d_in[base + 10];
        const float* bfc2  = (const float*)d_in[base + 11];

        ng_wconv_t<<<dim3(D / 64, 3 * D / 64), 256, 0, stream>>>(Wqkv, wq, D, 3 * D);
        ng_wconv_t<<<dim3(D / 64, D / 64), 256, 0, stream>>>(Wproj, wp, D, D);
        ng_wconv_t<<<dim3(D / 64, 4 * D / 64), 256, 0, stream>>>(Wfc, wf, D, 4 * D);
        ng_wconv_t<<<dim3(4 * D / 64, D / 64), 256, 0, stream>>>(Wfc2, w2, 4 * D, D);

        ng_ln<<<M, 256, 0, stream>>>(x, ln1_w, ln1_b, h16);
        ng_gemm<<<dim3(M / 128, (3 * D) / 128), 256, 0, stream>>>(
            h16, wq, bqkv, nullptr, qkv16, M, 3 * D, D, 0, 1);
        ng_attn<<<dim3(NG_T / 64, NG_B * NG_H), 256, 0, stream>>>(qkv16, yat16);
        ng_gemm<<<dim3(M / 128, D / 128), 256, 0, stream>>>(
            yat16, wp, bproj, x, x, M, D, D, 0, 0);
        ng_ln<<<M, 256, 0, stream>>>(x, ln2_w, ln2_b, h16);
        ng_gemm<<<dim3(M / 128, (4 * D) / 128), 256, 0, stream>>>(
            h16, wf, bfc, nullptr, fcb16, M, 4 * D, D, 1, 1);
        ng_gemm<<<dim3(M / 128, D / 128), 256, 0, stream>>>(
            fcb16, w2, bfc2, x, x, M, D, 4 * D, 0, 0);
    }

    ng_ln<<<M, 256, 0, stream>>>(x, lnf_w, lnf_b, h16);
    ng_gemm<<<dim3(M / 128, NG_V / 128), 256, 0, stream>>>(
        h16, tok16, nullptr, nullptr, (float*)d_out, M, NG_V, D, 0, 0);
}
